// IndustrialEyeSNN_67765993996473
// MI455X (gfx1250) — compile-verified, matrix operand order checked by reading, not run on hardware
//
#include <hip/hip_runtime.h>
#include <hip/hip_bf16.h>

typedef __attribute__((ext_vector_type(16))) _Float16 v16h;
typedef __attribute__((ext_vector_type(8)))  float    v8f;

// ---------------------------------------------------------------------------
// WMMA fragment helpers (CDNA5 V_WMMA_F32_16X16X32_F16 layouts, wave32)
// A (16xK row-major, lda): lane<16 -> K 0..7 / 16..23 ; lane>=16 -> 8..15 / 24..31
// B swizzled storage: tile = kstep*4 + ntile; element (kk,n) of a 32x16 tile
// lives at [(tile*32 + lane)*16 + i] with lane = ((kk>>3)&1)<<4 | (n&15),
// i = (kk&7) | ((kk>>4)<<3)  -> each lane's fragment is 16 contiguous halves.
// C/D f32: vgpr r, lane l -> (M = r + 8*(l>>4), N = l&15)
// ---------------------------------------------------------------------------
__device__ __forceinline__ v16h load_a16(const _Float16* A, int lda, int m0, int k0) {
  int lane = threadIdx.x & 31;
  const _Float16* row = A + (size_t)(m0 + (lane & 15)) * lda + k0 + ((lane >> 4) << 3);
  v16h a;
#pragma unroll
  for (int i = 0; i < 8; ++i) { a[i] = row[i]; a[i + 8] = row[16 + i]; }
  return a;
}

__device__ __forceinline__ v16h load_b16_swz(const _Float16* Bs, int tileIdx) {
  int lane = threadIdx.x & 31;
  return *(const v16h*)(Bs + ((size_t)tileIdx * 32 + lane) * 16);
}

__device__ __forceinline__ size_t b_swz_idx(int k, int n) {
  int kstep = k >> 5, kk = k & 31;
  int hi = (kk >> 3) & 1;
  int ii = (kk & 7) | ((kk >> 4) << 3);
  int lane = (hi << 4) | (n & 15);
  int ntile = n >> 4;
  return (((size_t)kstep * 4 + ntile) * 32 + lane) * 16 + ii;
}

// ---------------------------------------------------------------------------
// Kernel 0: zero P accumulator (512x64 f32) + spike counter
// ---------------------------------------------------------------------------
__global__ void snn_init(float* __restrict__ P, float* __restrict__ cnt) {
  int i = blockIdx.x * 256 + threadIdx.x;
  if (i < 32768) P[i] = 0.0f;
  if (i == 0) *cnt = 0.0f;
}

// ---------------------------------------------------------------------------
// Kernel 1: conv1 (2->16ch, 3x3, s2, p1) fused with LIF scan 1.
// One thread per (b, h1, w1); 16-channel mem/thresh state in registers,
// sequential loop over T=8. Writes s1 spikes as f16 (exact 0/1).
// s1 layout: [(b*8+t)*16 + ch][4096] (h1*64+w1) -> coalesced per-channel.
// ---------------------------------------------------------------------------
__global__ void __launch_bounds__(256)
snn_conv1_lif1(const float* __restrict__ x,
               const float* __restrict__ w1,
               const float* __restrict__ b1,
               const float* __restrict__ beta1,
               _Float16* __restrict__ s1) {
  __shared__ float wl[288];
  __shared__ float bl[16];
  __shared__ float betal[16];
  for (int i = threadIdx.x; i < 288; i += 256) wl[i] = w1[i];
  if (threadIdx.x < 16) { bl[threadIdx.x] = b1[threadIdx.x]; betal[threadIdx.x] = beta1[threadIdx.x]; }
  __syncthreads();

  int tid = blockIdx.x * 256 + threadIdx.x;   // 0 .. 262143
  int b   = tid >> 12;                        // /4096
  int pos = tid & 4095;
  int h1  = pos >> 6, ww = pos & 63;

  float mem[16], bth[16];
#pragma unroll
  for (int c = 0; c < 16; ++c) { mem[c] = 0.f; bth[c] = 0.f; }

  for (int t = 0; t < 8; ++t) {
    float xin[18];
#pragma unroll
    for (int ic = 0; ic < 2; ++ic)
#pragma unroll
      for (int r = 0; r < 3; ++r) {
        int ih = h1 * 2 + r - 1;
#pragma unroll
        for (int cc = 0; cc < 3; ++cc) {
          int iw = ww * 2 + cc - 1;
          float v = 0.f;
          if (ih >= 0 && ih < 128 && iw >= 0 && iw < 128)
            v = x[(((size_t)(b * 8 + t) * 2 + ic) * 128 + ih) * 128 + iw];
          xin[ic * 9 + r * 3 + cc] = v;
        }
      }
#pragma unroll
    for (int c = 0; c < 16; ++c) {
      float acc = bl[c];
#pragma unroll
      for (int k = 0; k < 18; ++k) acc = fmaf(wl[c * 18 + k], xin[k], acc);
      float m   = betal[c] * mem[c] + acc;
      float th  = 1.0f + bth[c];
      float spk = (m - th >= 0.f) ? 1.f : 0.f;
      mem[c] = m - spk * th;
      bth[c] = 0.9f * bth[c] + 0.1f * spk;
      s1[((size_t)(b * 8 + t) * 16 + c) * 4096 + pos] = (_Float16)spk;
    }
  }
}

// ---------------------------------------------------------------------------
// Kernel 2: conv2 (16->32ch, 3x3, s2, p1) as WMMA GEMM fused with LIF scan 2.
// M=32 channels, K=144 (pad 160), N=64 positions/block; 8 waves = 2Mx4N tiles.
// A fragments are loaded from LDS right before each WMMA (2x ds_load_b128 —
// preloading them across the t-loop caused scratch spills in the previous
// build). Patch LDS is written fragment-swizzled so B loads are also
// 2x ds_load_b128. Dual accumulators break the WMMA->WMMA RAW chain.
// ---------------------------------------------------------------------------
__global__ void __launch_bounds__(256)
snn_conv2_lif2(const _Float16* __restrict__ s1,
               const float* __restrict__ cw2,
               const float* __restrict__ cb2,
               const float* __restrict__ beta2,
               _Float16* __restrict__ feats,
               float* __restrict__ count) {
  __shared__ alignas(32) _Float16 al[32 * 160];        // weights (row-major, K pad 160)
  __shared__ alignas(32) _Float16 pl[5 * 4 * 32 * 16]; // patches, frag-swizzled

  int b    = blockIdx.x >> 4;         // 64 batches
  int tile = blockIdx.x & 15;         // 16 tiles of 64 positions

  for (int i = threadIdx.x; i < 32 * 160; i += 256) {
    int m = i / 160, k = i % 160;
    al[i] = (k < 144) ? (_Float16)cw2[m * 144 + k] : (_Float16)0.f;
  }

  int lane = threadIdx.x & 31;
  int wid  = threadIdx.x >> 5;        // 0..7
  int m0   = (wid & 1) * 16;
  int ntw  = (wid >> 1);              // n-tile 0..3
  int hi   = lane >> 4;
  int pos  = tile * 64 + ntw * 16 + (lane & 15);   // output spatial index

  float betav[8], biasv[8];
  int   offv[8];                       // feats store offset: ch*1024 + pos
#pragma unroll
  for (int r = 0; r < 8; ++r) {
    int ch   = m0 + r + 8 * hi;
    betav[r] = beta2[ch];
    biasv[r] = cb2[ch];
    offv[r]  = ch * 1024 + pos;
  }

  // Fill-loop invariants: with 256 threads and 64 columns, each thread's n
  // is fixed and k advances by 4 per iteration.
  int nfill = threadIdx.x & 63;
  int kbase = threadIdx.x >> 6;                  // 0..3
  int pfill = tile * 64 + nfill;
  int h2f   = pfill >> 5, w2f = pfill & 31;

  v8f mem = {};
  v8f bth = {};
  float cnt = 0.f;

  for (int t = 0; t < 8; ++t) {
    __syncthreads();   // previous iteration's fragment loads complete
    const _Float16* s1b = s1 + (size_t)(b * 8 + t) * 16 * 4096;
#pragma unroll
    for (int j = 0; j < 40; ++j) {
      int k = kbase + j * 4;                     // 0..159
      _Float16 v = (_Float16)0.f;
      if (k < 144) {
        int ic = k / 9, rr = (k % 9) / 3, cc = k % 3;
        int ih = h2f * 2 + rr - 1, iw = w2f * 2 + cc - 1;
        if (ih >= 0 && ih < 64 && iw >= 0 && iw < 64)
          v = s1b[ic * 4096 + ih * 64 + iw];
      }
      pl[b_swz_idx(k, nfill)] = v;
    }
    __syncthreads();

    v8f ca, cb = {};
#pragma unroll
    for (int r = 0; r < 8; ++r) ca[r] = betav[r] * mem[r];   // seed beta*mem

    {
      v16h a0 = load_a16(al, 160, m0, 0);
      v16h b0 = load_b16_swz(pl, 0 * 4 + ntw);
      ca = __builtin_amdgcn_wmma_f32_16x16x32_f16(false, a0, false, b0, (short)0, ca, false, false);
      v16h a1 = load_a16(al, 160, m0, 32);
      v16h b1 = load_b16_swz(pl, 1 * 4 + ntw);
      cb = __builtin_amdgcn_wmma_f32_16x16x32_f16(false, a1, false, b1, (short)0, cb, false, false);
      v16h a2 = load_a16(al, 160, m0, 64);
      v16h b2 = load_b16_swz(pl, 2 * 4 + ntw);
      ca = __builtin_amdgcn_wmma_f32_16x16x32_f16(false, a2, false, b2, (short)0, ca, false, false);
      v16h a3 = load_a16(al, 160, m0, 96);
      v16h b3 = load_b16_swz(pl, 3 * 4 + ntw);
      cb = __builtin_amdgcn_wmma_f32_16x16x32_f16(false, a3, false, b3, (short)0, cb, false, false);
      v16h a4 = load_a16(al, 160, m0, 128);
      v16h b4 = load_b16_swz(pl, 4 * 4 + ntw);
      ca = __builtin_amdgcn_wmma_f32_16x16x32_f16(false, a4, false, b4, (short)0, ca, false, false);
    }

    _Float16* fbase = feats + (size_t)(b * 8 + t) * 32768;
#pragma unroll
    for (int r = 0; r < 8; ++r) {
      float m   = ca[r] + cb[r] + biasv[r];
      float th  = 1.0f + bth[r];
      float spk = (m - th >= 0.f) ? 1.f : 0.f;
      mem[r] = m - spk * th;
      bth[r] = 0.9f * bth[r] + 0.1f * spk;
      cnt += spk;
      fbase[offv[r]] = (_Float16)spk;
    }
  }
  atomicAdd(count, cnt);
}

// ---------------------------------------------------------------------------
// Kernel 3: convert proj_w (64 x 32768 f32) -> pwT f16 in fragment-swizzled
// layout (1024 k-steps x 4 n-tiles x 32 lanes x 16 halves).
// ---------------------------------------------------------------------------
__global__ void __launch_bounds__(256)
snn_cvt_pw(const float* __restrict__ pw, _Float16* __restrict__ pwT) {
  int i = blockIdx.x * 256 + threadIdx.x;       // 0 .. 2097151
  int n = i >> 15;
  int k = i & 32767;
  pwT[b_swz_idx(k, n)] = (_Float16)pw[i];
}

// ---------------------------------------------------------------------------
// Kernel 4: proj GEMM (512 x 32768) @ (32768 x 64), WMMA f16, f32 accum.
// Grid = 32 M-tiles x 16 K-chunks (2048 K each); 64 threads = 2 waves, each
// wave owns 2 N-tiles sharing one A fragment per k-step. Partials reduced
// into P with f32 atomics (P pre-zeroed; proj_b applied by consumer).
// ---------------------------------------------------------------------------
__global__ void __launch_bounds__(64)
snn_proj_gemm(const _Float16* __restrict__ feats,
              const _Float16* __restrict__ pwT,
              float* __restrict__ P) {
  int m0   = (blockIdx.x >> 4) * 16;   // 32 m-tiles
  int kc   = blockIdx.x & 15;          // 16 k-chunks
  int wid  = threadIdx.x >> 5;         // 0..1
  int lane = threadIdx.x & 31;
  int ntA  = wid * 2;                  // n-tiles {0,1} or {2,3}

  v8f c0 = {};
  v8f c1 = {};
  int ksBeg = kc * 64;
#pragma unroll 2
  for (int ks = ksBeg; ks < ksBeg + 64; ++ks) {
    v16h a  = load_a16(feats, 32768, m0, ks * 32);
    v16h b0 = load_b16_swz(pwT, ks * 4 + ntA);
    v16h b1 = load_b16_swz(pwT, ks * 4 + ntA + 1);
    c0 = __builtin_amdgcn_wmma_f32_16x16x32_f16(false, a, false, b0, (short)0, c0, false, false);
    c1 = __builtin_amdgcn_wmma_f32_16x16x32_f16(false, a, false, b1, (short)0, c1, false, false);
  }
  int col = lane & 15, hi = lane >> 4;
  int n0  = ntA * 16;
#pragma unroll
  for (int r = 0; r < 8; ++r) {
    int row = m0 + r + 8 * hi;
    atomicAdd(&P[row * 64 + n0 + col],      c0[r]);
    atomicAdd(&P[row * 64 + n0 + 16 + col], c1[r]);
  }
}

// ---------------------------------------------------------------------------
// Kernel 5: per-batch MHA (4 heads, dh=16, T=8) + mean + output spike + logits
// One 64-thread block per batch element; proj_b folded into the P load.
// ---------------------------------------------------------------------------
__global__ void __launch_bounds__(64)
snn_attn_head(const float* __restrict__ P,
              const float* __restrict__ pb,
              const float* __restrict__ wq,
              const float* __restrict__ wk,
              const float* __restrict__ wv,
              const float* __restrict__ wo,
              const float* __restrict__ clsw,
              const float* __restrict__ clsb,
              float* __restrict__ out) {
  __shared__ float Pl[8][64], ql[8][64], kl[8][64], vl[8][64], ol[8][64], spkl[64];
  int b = blockIdx.x;
  int d = threadIdx.x;   // 0..63

  float biasd = pb[d];
  for (int t = 0; t < 8; ++t) Pl[t][d] = P[(b * 8 + t) * 64 + d] + biasd;
  __syncthreads();

  for (int t = 0; t < 8; ++t) {
    float aq = 0.f, ak = 0.f, av = 0.f;
#pragma unroll 8
    for (int j = 0; j < 64; ++j) {
      float pv = Pl[t][j];
      aq = fmaf(pv, wq[d * 64 + j], aq);
      ak = fmaf(pv, wk[d * 64 + j], ak);
      av = fmaf(pv, wv[d * 64 + j], av);
    }
    ql[t][d] = aq; kl[t][d] = ak; vl[t][d] = av;
  }
  __syncthreads();

  if (threadIdx.x < 32) {
    int h  = threadIdx.x >> 3;   // head 0..3
    int tq = threadIdx.x & 7;
    float att[8];
    float mx = -1e30f;
    for (int tk = 0; tk < 8; ++tk) {
      float a = 0.f;
#pragma unroll
      for (int dd = 0; dd < 16; ++dd)
        a = fmaf(ql[tq][h * 16 + dd], kl[tk][h * 16 + dd], a);
      a *= 0.25f;   // dh^-0.5, dh = 16
      att[tk] = a;
      mx = fmaxf(mx, a);
    }
    float s = 0.f;
    for (int tk = 0; tk < 8; ++tk) { att[tk] = __expf(att[tk] - mx); s += att[tk]; }
    float inv = 1.f / s;
#pragma unroll
    for (int dd = 0; dd < 16; ++dd) {
      float o = 0.f;
      for (int tk = 0; tk < 8; ++tk) o = fmaf(att[tk] * inv, vl[tk][h * 16 + dd], o);
      ol[tq][h * 16 + dd] = o;
    }
  }
  __syncthreads();

  {
    float ctx = 0.f;
    for (int tq = 0; tq < 8; ++tq) {
      float a = 0.f;
#pragma unroll 8
      for (int j = 0; j < 64; ++j) a = fmaf(ol[tq][j], wo[d * 64 + j], a);
      ctx += a;
    }
    ctx *= 0.125f;                      // mean over T
    spkl[d] = (ctx - 1.0f >= 0.f) ? 1.f : 0.f;
  }
  __syncthreads();

  if (d < 2) {
    float acc = clsb[d];
#pragma unroll 8
    for (int j = 0; j < 64; ++j) acc = fmaf(spkl[j], clsw[d * 64 + j], acc);
    out[b * 2 + d] = acc;
  }
}

// ---------------------------------------------------------------------------
// Kernel 6: sparsity = 1 - total_spikes / (512*32768)
// ---------------------------------------------------------------------------
__global__ void snn_final(const float* __restrict__ cnt, float* __restrict__ out) {
  if (threadIdx.x == 0 && blockIdx.x == 0)
    out[128] = 1.0f - (*cnt) / (512.0f * 32768.0f);
}

// ---------------------------------------------------------------------------
// Host launch
// ---------------------------------------------------------------------------
extern "C" void kernel_launch(void* const* d_in, const int* in_sizes, int n_in,
                              void* d_out, int out_size, void* d_ws, size_t ws_size,
                              hipStream_t stream) {
  const float* x     = (const float*)d_in[0];
  const float* c1w   = (const float*)d_in[1];
  const float* c1b   = (const float*)d_in[2];
  const float* c2w   = (const float*)d_in[3];
  const float* c2b   = (const float*)d_in[4];
  const float* beta1 = (const float*)d_in[5];
  const float* beta2 = (const float*)d_in[6];
  const float* pw    = (const float*)d_in[7];
  const float* pb    = (const float*)d_in[8];
  const float* wq    = (const float*)d_in[9];
  const float* wk    = (const float*)d_in[10];
  const float* wv    = (const float*)d_in[11];
  const float* wo    = (const float*)d_in[12];
  const float* clsw  = (const float*)d_in[13];
  const float* clsb  = (const float*)d_in[14];
  float* out = (float*)d_out;

  // Workspace layout (bytes):
  //   s1    f16 [512][16][4096]      :       0 ..  67108864
  //   feats f16 [512][32768]         : 67108864 .. 100663296
  //   pwT   f16 swizzled [2097152]   : 100663296 .. 104857600
  //   P     f32 [512][64]            : 104857600 .. 104988672
  //   count f32 [1]                  : 104988672
  char* ws = (char*)d_ws;
  _Float16* s1    = (_Float16*)(ws);
  _Float16* feats = (_Float16*)(ws + 67108864);
  _Float16* pwT   = (_Float16*)(ws + 100663296);
  float*    P     = (float*)(ws + 104857600);
  float*    cnt   = (float*)(ws + 104988672);

  snn_init<<<128, 256, 0, stream>>>(P, cnt);
  snn_conv1_lif1<<<1024, 256, 0, stream>>>(x, c1w, c1b, beta1, s1);
  snn_conv2_lif2<<<1024, 256, 0, stream>>>(s1, c2w, c2b, beta2, feats, cnt);
  snn_cvt_pw<<<8192, 256, 0, stream>>>(pw, pwT);
  snn_proj_gemm<<<512, 64, 0, stream>>>(feats, pwT, P);
  snn_attn_head<<<64, 64, 0, stream>>>(P, pb, wq, wk, wv, wo, clsw, clsb, out);
  snn_final<<<1, 32, 0, stream>>>(cnt, out);
}
